// DepthHead_56985626083916
// MI455X (gfx1250) — compile-verified, matrix operand order checked by reading, not run on hardware
//
#include <hip/hip_runtime.h>
#include <hip/hip_bf16.h>

// ---------------------------------------------------------------- types ----
typedef _Float16 v16h __attribute__((ext_vector_type(16)));
typedef _Float16 v8h  __attribute__((ext_vector_type(8)));
typedef float    v8f  __attribute__((ext_vector_type(8)));
typedef unsigned int u32x4 __attribute__((ext_vector_type(4)));
typedef int          i32x4 __attribute__((ext_vector_type(4)));
typedef int          i32x8 __attribute__((ext_vector_type(8)));

#define BATCH 32
#define NPTS  4096
#define BN    (BATCH * NPTS)   // 131072
#define KDIM  1024
#define EPSF  1e-5f

#if defined(__gfx1250__) && __has_builtin(__builtin_amdgcn_tensor_load_to_lds) && __has_builtin(__builtin_amdgcn_s_wait_tensorcnt)
#define USE_TDM 1
#endif

// ----------------------------------------------------- TDM 2D tile load ----
// Loads a tile of (tile_h rows) x (tile_w f16 elements) from global memory
// (row stride row_stride_elems) into LDS at lds_dst. Descriptor per
// CDNA5 ISA (D# group0/group1, 2D tensor, data_size=2B).
// Issue-only: caller is responsible for s_wait_tensorcnt.
__device__ inline void tdm_load_2d_f16(_Float16* lds_dst, const _Float16* gsrc,
                                       unsigned tile_w, unsigned tile_h,
                                       unsigned long long row_stride_elems) {
#ifdef USE_TDM
  unsigned long long ga = (unsigned long long)(const void*)gsrc;
  unsigned lds = (unsigned)(unsigned long long)(void*)lds_dst;  // LDS aperture low bits
  u32x4 g0;
  g0[0] = 1u;                                              // count=1, user desc
  g0[1] = lds;                                             // lds_addr
  g0[2] = (unsigned)(ga & 0xFFFFFFFFu);                    // global_addr lo
  g0[3] = (unsigned)((ga >> 32) & 0x01FFFFFFu) | (2u << 30); // global_addr hi | type=2
  unsigned long long td0 = 1u << 30;                       // huge tensor dims: no clipping
  unsigned long long td1 = 1u << 30;
  unsigned long long st0 = row_stride_elems;
  i32x8 g1;
  g1[0] = (int)(1u << 16);                                 // data_size=1 (2 bytes)
  g1[1] = (int)((td0 & 0xFFFFu) << 16);                    // tensor_dim0[15:0]
  g1[2] = (int)(((td0 >> 16) & 0xFFFFu) | ((td1 & 0xFFFFu) << 16));
  g1[3] = (int)(((td1 >> 16) & 0xFFFFu) | (tile_w << 16)); // tile_dim0
  g1[4] = (int)(tile_h & 0xFFFFu);                         // tile_dim1
  g1[5] = (int)(st0 & 0xFFFFFFFFu);                        // tensor_dim0_stride lo
  g1[6] = (int)((st0 >> 32) & 0xFFFFu);                    // tensor_dim0_stride hi
  g1[7] = 0;
  i32x4 gz = {0, 0, 0, 0};
#if __clang_major__ >= 23
  i32x8 gz8 = {0, 0, 0, 0, 0, 0, 0, 0};
  __builtin_amdgcn_tensor_load_to_lds(g0, g1, gz, gz, gz8, 0);
#else
  __builtin_amdgcn_tensor_load_to_lds(g0, g1, gz, gz, 0);
#endif
#endif
}

// --------------------------------------------------------------- zeroing ---
__global__ void zero_kernel(float* p, int n) {
  int i = blockIdx.x * blockDim.x + threadIdx.x;
  if (i < n) p[i] = 0.f;
}

// --------------------------------------------- small res-MLP layer (VALU) --
// One block per batch. Two passes inside the kernel: (1) y = xW+b, s = xWs+bs,
// accumulate per-channel sum/sumsq over N via wave32 shuffle + LDS atomics,
// (2) z = relu((y-mu)*rstd) + s.  ybuf rows hold [y(64) | s(64)].
template <int CIN, int COUT>
__global__ __launch_bounds__(1024) void small_layer_kernel(
    const float* __restrict__ x, int xstride,
    const float* __restrict__ W, const float* __restrict__ b,
    const float* __restrict__ Ws, const float* __restrict__ bs,
    float* __restrict__ ybuf, float* __restrict__ zout) {
  __shared__ float sW[CIN * COUT];
  __shared__ float sWs[CIN * COUT];
  __shared__ float sb[COUT], sbs[COUT];
  __shared__ float ssum[COUT], sssq[COUT];
  __shared__ float smu[COUT], srs[COUT];
  int bi = blockIdx.x, tid = threadIdx.x;
  for (int i = tid; i < CIN * COUT; i += 1024) { sW[i] = W[i]; sWs[i] = Ws[i]; }
  for (int i = tid; i < COUT; i += 1024) { sb[i] = b[i]; sbs[i] = bs[i]; ssum[i] = 0.f; sssq[i] = 0.f; }
  __syncthreads();
  for (int chunk = 0; chunk < NPTS / 1024; ++chunk) {
    int n = chunk * 1024 + tid;
    size_t r = (size_t)bi * NPTS + n;
    float xv[CIN];
#pragma unroll
    for (int k = 0; k < CIN; k++) xv[k] = x[r * xstride + k];
#pragma unroll
    for (int c = 0; c < COUT; c++) {
      float y = sb[c], s = sbs[c];
#pragma unroll
      for (int k = 0; k < CIN; k++) { y += xv[k] * sW[k * COUT + c]; s += xv[k] * sWs[k * COUT + c]; }
      ybuf[r * 128 + c] = y;
      ybuf[r * 128 + 64 + c] = s;
      float a = y, q = y * y;
      for (int o = 16; o > 0; o >>= 1) { a += __shfl_down(a, o, 32); q += __shfl_down(q, o, 32); }
      if ((tid & 31) == 0) { atomicAdd(&ssum[c], a); atomicAdd(&sssq[c], q); }
    }
  }
  __syncthreads();
  for (int c = tid; c < COUT; c += 1024) {
    float mu = ssum[c] * (1.f / NPTS);
    float var = sssq[c] * (1.f / NPTS) - mu * mu;
    smu[c] = mu;
    srs[c] = rsqrtf(var + EPSF);
  }
  __syncthreads();
  for (int chunk = 0; chunk < NPTS / 1024; ++chunk) {
    int n = chunk * 1024 + tid;
    size_t r = (size_t)bi * NPTS + n;
#pragma unroll
    for (int c = 0; c < COUT; c++) {
      float y = ybuf[r * 128 + c], s = ybuf[r * 128 + 64 + c];
      float t = (y - smu[c]) * srs[c];
      zout[r * 64 + c] = fmaxf(t, 0.f) + s;
    }
  }
}

// ------------------------------------------------- bilinear grid sampling --
template <int C, int H, int Wd>
__device__ inline void sample_map(const float* __restrict__ f, int b,
                                  float gxn, float gyn, _Float16* __restrict__ out) {
  float gx = (gxn + 1.f) * (Wd * 0.5f) - 0.5f;
  float gy = (gyn + 1.f) * (H * 0.5f) - 0.5f;
  float x0f = floorf(gx), y0f = floorf(gy);
  int x0 = (int)x0f, y0 = (int)y0f;
  float wx1 = gx - x0f, wy1 = gy - y0f;
  float wx0 = 1.f - wx1, wy0 = 1.f - wy1;
  float vx0 = (x0 >= 0 && x0 < Wd) ? 1.f : 0.f;
  float vx1 = (x0 + 1 >= 0 && x0 + 1 < Wd) ? 1.f : 0.f;
  float vy0 = (y0 >= 0 && y0 < H) ? 1.f : 0.f;
  float vy1 = (y0 + 1 >= 0 && y0 + 1 < H) ? 1.f : 0.f;
  int xc0 = x0 < 0 ? 0 : (x0 > Wd - 1 ? Wd - 1 : x0);
  int xc1 = x0 + 1 < 0 ? 0 : (x0 + 1 > Wd - 1 ? Wd - 1 : x0 + 1);
  int yc0 = y0 < 0 ? 0 : (y0 > H - 1 ? H - 1 : y0);
  int yc1 = y0 + 1 < 0 ? 0 : (y0 + 1 > H - 1 ? H - 1 : y0 + 1);
  float w00 = wx0 * wy0 * vx0 * vy0, w10 = wx1 * wy0 * vx1 * vy0;
  float w01 = wx0 * wy1 * vx0 * vy1, w11 = wx1 * wy1 * vx1 * vy1;
  size_t base = (size_t)b * C * H * Wd;
  int i00 = yc0 * Wd + xc0, i10 = yc0 * Wd + xc1;
  int i01 = yc1 * Wd + xc0, i11 = yc1 * Wd + xc1;
#pragma unroll 4
  for (int c = 0; c < C; c++) {
    size_t o = base + (size_t)c * (H * Wd);
    float v = f[o + i00] * w00 + f[o + i10] * w10 + f[o + i01] * w01 + f[o + i11] * w11;
    out[c] = (_Float16)v;
  }
}

__global__ __launch_bounds__(256) void gather_kernel(
    const float* __restrict__ f0, const float* __restrict__ f1,
    const float* __restrict__ f2, const float* __restrict__ f3,
    const float* __restrict__ WH, const float* __restrict__ zenc,
    _Float16* __restrict__ feats) {
  int p = blockIdx.x * 256 + threadIdx.x;
  if (p >= BN) return;
  int b = p >> 12;
  float gx = WH[(size_t)p * 2 + 0], gy = WH[(size_t)p * 2 + 1];
  __builtin_prefetch(zenc + (size_t)p * 64, 0, 0);  // global_prefetch_b8
  _Float16* row = feats + (size_t)p * KDIM;
  sample_map<512, 16, 16>(f0, b, gx, gy, row);
  sample_map<256, 32, 32>(f1, b, gx, gy, row + 512);
  sample_map<128, 64, 64>(f2, b, gx, gy, row + 768);
  sample_map<64, 128, 128>(f3, b, gx, gy, row + 896);
#pragma unroll
  for (int c = 0; c < 64; c++) row[960 + c] = (_Float16)zenc[(size_t)p * 64 + c];
}

// --------------------------- pack dec1 weights to f16, TRANSPOSED ---------
// wcat_t[n][k] ([128][1024]); col-major panels become the same 2D tile shape
// as the activation panels, so the TDM can stage both.
__global__ void pack_w_kernel(const float* __restrict__ W, const float* __restrict__ b,
                              const float* __restrict__ Ws, const float* __restrict__ bs,
                              _Float16* __restrict__ wcat_t, float* __restrict__ bcat) {
  int i = blockIdx.x * blockDim.x + threadIdx.x;  // 0 .. 128*1024
  if (i < 128 * KDIM) {
    int n = i >> 10, k = i & 1023;
    float v = (n < 64) ? W[k * 64 + n] : Ws[k * 64 + (n - 64)];
    wcat_t[i] = (_Float16)v;
  }
  if (i < 128) bcat[i] = (i < 64) ? b[i] : bs[i - 64];
}

// ---------------------------------------------------- dec1 WMMA GEMM ------
// Y[131072,128] = feats[131072,1024](f16) @ W[1024,128](f16, stored
// transposed) + bcat, f32 accumulate via v_wmma_f32_16x16x32_f16.
// Both LDS panels are double-buffered and staged by the Tensor Data Mover;
// the next panel's TDM is issued before computing the current one so the DMA
// overlaps the WMMA chain. Per-(b,c<64) sum/sumsq fused for InstanceNorm.
__global__ __launch_bounds__(256) void gemm_wmma_kernel(
    const _Float16* __restrict__ A, const _Float16* __restrict__ Bt,
    const float* __restrict__ bcat, float* __restrict__ ybuf,
    float* __restrict__ stats) {
  __shared__ _Float16 sA[2][128][32];   // A panels: 128 rows x 32 K  (2 x 8KB)
  __shared__ _Float16 sBt[2][128][32];  // B panels: 128 cols x 32 K  (2 x 8KB)
  __shared__ float ssum[64], sssq[64];
  int tid = threadIdx.x;
  int wave = tid >> 5, lane = tid & 31;
  int lr = lane & 15, hi = lane >> 4;
  size_t row0 = (size_t)blockIdx.x * 128;
  int bb = (int)(row0 >> 12);  // batch of this row tile (128 | 4096)
  if (tid < 64) { ssum[tid] = 0.f; sssq[tid] = 0.f; }
  v8f acc[8];
  v8f zv = {0.f, 0.f, 0.f, 0.f, 0.f, 0.f, 0.f, 0.f};
#pragma unroll
  for (int t = 0; t < 8; t++) acc[t] = zv;

#ifdef USE_TDM
  if (wave == 0) {  // prologue: stage panel 0 via TDM
    tdm_load_2d_f16(&sA[0][0][0], A + row0 * KDIM, 32u, 128u, (unsigned long long)KDIM);
    tdm_load_2d_f16(&sBt[0][0][0], Bt, 32u, 128u, (unsigned long long)KDIM);
    __builtin_amdgcn_s_wait_tensorcnt(0);
  }
#else
  {
    int rowa = tid >> 1, colh = (tid & 1) * 16;
    const _Float16* srca = A + (row0 + rowa) * KDIM + colh;
    *(v8h*)&sA[0][rowa][colh] = *(const v8h*)srca;
    *(v8h*)&sA[0][rowa][colh + 8] = *(const v8h*)(srca + 8);
    const _Float16* srcb = Bt + (size_t)rowa * KDIM + colh;
    *(v8h*)&sBt[0][rowa][colh] = *(const v8h*)srcb;
    *(v8h*)&sBt[0][rowa][colh + 8] = *(const v8h*)(srcb + 8);
  }
#endif
  __syncthreads();

#pragma unroll 2
  for (int k0 = 0; k0 < KDIM; k0 += 32) {
    int buf = (k0 >> 5) & 1;
    int nxt = buf ^ 1;
    bool has_next = (k0 + 32) < KDIM;

    // issue next panel loads before computing the current one (overlap)
#ifdef USE_TDM
    if (has_next && wave == 0) {
      tdm_load_2d_f16(&sA[nxt][0][0], A + row0 * KDIM + (k0 + 32), 32u, 128u,
                      (unsigned long long)KDIM);
      tdm_load_2d_f16(&sBt[nxt][0][0], Bt + (k0 + 32), 32u, 128u,
                      (unsigned long long)KDIM);
    }
#else
    if (has_next) {
      int rowa = tid >> 1, colh = (tid & 1) * 16;
      const _Float16* srca = A + (row0 + rowa) * KDIM + (k0 + 32) + colh;
      *(v8h*)&sA[nxt][rowa][colh] = *(const v8h*)srca;
      *(v8h*)&sA[nxt][rowa][colh + 8] = *(const v8h*)(srca + 8);
      const _Float16* srcb = Bt + (size_t)rowa * KDIM + (k0 + 32) + colh;
      *(v8h*)&sBt[nxt][rowa][colh] = *(const v8h*)srcb;
      *(v8h*)&sBt[nxt][rowa][colh + 8] = *(const v8h*)(srcb + 8);
    }
#endif

    // A fragment: lanes 0-15 hold M rows with K{0..7,16..23}, lanes 16-31 K{8..15,24..31}
    int arow = wave * 16 + lr;
    int kb = hi * 8;
    v8h a0 = *(const v8h*)&sA[buf][arow][kb];
    v8h a1 = *(const v8h*)&sA[buf][arow][kb + 16];
    v16h af;
#pragma unroll
    for (int i = 0; i < 8; i++) { af[i] = a0[i]; af[8 + i] = a1[i]; }

    // preload all 8 B fragments so the ds_loads clause and WMMAs go back-to-back
    v16h bfs[8];
#pragma unroll
    for (int t = 0; t < 8; t++) {
      int col = t * 16 + lr;
      v8h b0 = *(const v8h*)&sBt[buf][col][kb];
      v8h b1 = *(const v8h*)&sBt[buf][col][kb + 16];
#pragma unroll
      for (int i = 0; i < 8; i++) { bfs[t][i] = b0[i]; bfs[t][8 + i] = b1[i]; }
    }
#pragma unroll
    for (int t = 0; t < 8; t++) {
      acc[t] = __builtin_amdgcn_wmma_f32_16x16x32_f16(
          false, af, false, bfs[t], (short)0, acc[t], false, false);
    }

#ifdef USE_TDM
    if (has_next && wave == 0) __builtin_amdgcn_s_wait_tensorcnt(0);
#endif
    __syncthreads();  // reads of buf done; writes (incl. TDM) to nxt visible
  }

  // Epilogue: C/D layout — lane<16: (M=v, N=lane); lane>=16: (M=v+8, N=lane-16)
#pragma unroll
  for (int t = 0; t < 8; t++) {
    int col = t * 16 + lr;
    float bv = bcat[col];
    float ps = 0.f, pq = 0.f;
#pragma unroll
    for (int v = 0; v < 8; v++) {
      int m = wave * 16 + hi * 8 + v;
      float y = acc[t][v] + bv;
      ybuf[(row0 + m) * 128 + col] = y;
      ps += y;
      pq += y * y;
    }
    if (col < 64) { atomicAdd(&ssum[col], ps); atomicAdd(&sssq[col], pq); }
  }
  __syncthreads();
  if (tid < 64) {
    atomicAdd(&stats[bb * 128 + tid], ssum[tid]);
    atomicAdd(&stats[bb * 128 + 64 + tid], sssq[tid]);
  }
}

// -------------------------------------------- dec1 InstanceNorm + shortcut -
__global__ void dec1_norm_kernel(const float* __restrict__ ybuf,
                                 const float* __restrict__ stats,
                                 float* __restrict__ zout) {
  int i = blockIdx.x * blockDim.x + threadIdx.x;  // BN*64
  if (i >= BN * 64) return;
  int r = i >> 6, c = i & 63;
  int b = r >> 12;
  float mu = stats[b * 128 + c] * (1.f / NPTS);
  float var = stats[b * 128 + 64 + c] * (1.f / NPTS) - mu * mu;
  float rs = rsqrtf(var + EPSF);
  float y = ybuf[(size_t)r * 128 + c], s = ybuf[(size_t)r * 128 + 64 + c];
  zout[(size_t)r * 64 + c] = fmaxf((y - mu) * rs, 0.f) + s;
}

// --------------------------- final linear 16->1, mean-center, assemble out -
__global__ __launch_bounds__(1024) void final_kernel(
    const float* __restrict__ z16, const float* __restrict__ WH,
    const float* __restrict__ gw, const float* __restrict__ gb,
    float* __restrict__ out, float* __restrict__ tmp) {
  __shared__ float stot;
  __shared__ float smean;
  int b = blockIdx.x, tid = threadIdx.x;
  if (tid == 0) stot = 0.f;
  __syncthreads();
  float w[16];
#pragma unroll
  for (int k = 0; k < 16; k++) w[k] = gw[k];
  float bias = gb[0];
  float lsum = 0.f;
  for (int chunk = 0; chunk < NPTS / 1024; ++chunk) {
    int n = chunk * 1024 + tid;
    size_t r = (size_t)b * NPTS + n;
    float v = bias;
#pragma unroll
    for (int k = 0; k < 16; k++) v += z16[r * 64 + k] * w[k];
    tmp[r] = v;
    lsum += v;
  }
  for (int o = 16; o > 0; o >>= 1) lsum += __shfl_down(lsum, o, 32);
  if ((tid & 31) == 0) atomicAdd(&stot, lsum);
  __syncthreads();
  if (tid == 0) smean = stot * (1.f / NPTS);
  __syncthreads();
  float mu = smean;
  for (int chunk = 0; chunk < NPTS / 1024; ++chunk) {
    int n = chunk * 1024 + tid;
    size_t r = (size_t)b * NPTS + n;
    out[r * 3 + 0] = WH[r * 2 + 0] * 0.75f;
    out[r * 3 + 1] = -WH[r * 2 + 1] * 0.75f;
    out[r * 3 + 2] = tmp[r] - mu;
  }
}

// ------------------------------------------------------------- launcher ----
extern "C" void kernel_launch(void* const* d_in, const int* in_sizes, int n_in,
                              void* d_out, int out_size, void* d_ws, size_t ws_size,
                              hipStream_t stream) {
  const float* f0 = (const float*)d_in[0];
  const float* f1 = (const float*)d_in[1];
  const float* f2 = (const float*)d_in[2];
  const float* f3 = (const float*)d_in[3];
  const float* WH = (const float*)d_in[4];
  const float* rp = (const float*)d_in[5];
  // params flattened in dict order: enc1,enc2,enc3,dec1,dec2,dec3 {W,b,Ws,bs}, genz_W, genz_b
  const float* e1W = (const float*)d_in[6];  const float* e1b = (const float*)d_in[7];
  const float* e1Ws = (const float*)d_in[8]; const float* e1bs = (const float*)d_in[9];
  const float* e2W = (const float*)d_in[10]; const float* e2b = (const float*)d_in[11];
  const float* e2Ws = (const float*)d_in[12]; const float* e2bs = (const float*)d_in[13];
  const float* e3W = (const float*)d_in[14]; const float* e3b = (const float*)d_in[15];
  const float* e3Ws = (const float*)d_in[16]; const float* e3bs = (const float*)d_in[17];
  const float* d1W = (const float*)d_in[18]; const float* d1b = (const float*)d_in[19];
  const float* d1Ws = (const float*)d_in[20]; const float* d1bs = (const float*)d_in[21];
  const float* d2W = (const float*)d_in[22]; const float* d2b = (const float*)d_in[23];
  const float* d2Ws = (const float*)d_in[24]; const float* d2bs = (const float*)d_in[25];
  const float* d3W = (const float*)d_in[26]; const float* d3b = (const float*)d_in[27];
  const float* d3Ws = (const float*)d_in[28]; const float* d3bs = (const float*)d_in[29];
  const float* gzW = (const float*)d_in[30]; const float* gzb = (const float*)d_in[31];
  float* out = (float*)d_out;

  char* ws = (char*)d_ws;
  float* zA = (float*)(ws);                                // [BN,64] f32 (32 MB)
  float* zB = (float*)(ws + ((size_t)32 << 20));           // [BN,64] f32 (32 MB)
  float* ybuf = (float*)(ws + ((size_t)64 << 20));         // [BN,128] f32 (64 MB)
  _Float16* feats = (_Float16*)(ws + ((size_t)128 << 20)); // [BN,1024] f16 (256 MB)
  _Float16* wcat_t = (_Float16*)(ws + ((size_t)384 << 20)); // [128,1024] f16 (transposed)
  float* bcat = (float*)(ws + ((size_t)385 << 20));        // [128] f32
  float* stats = (float*)(ws + ((size_t)385 << 20) + 4096); // [B,128] f32

  // stats must be zero each call (graph-replay deterministic)
  zero_kernel<<<(BATCH * 128 + 255) / 256, 256, 0, stream>>>(stats, BATCH * 128);

  // encoder: 1 -> 16 -> 32 -> 64
  small_layer_kernel<1, 16><<<BATCH, 1024, 0, stream>>>(rp, 1, e1W, e1b, e1Ws, e1bs, ybuf, zA);
  small_layer_kernel<16, 32><<<BATCH, 1024, 0, stream>>>(zA, 64, e2W, e2b, e2Ws, e2bs, ybuf, zB);
  small_layer_kernel<32, 64><<<BATCH, 1024, 0, stream>>>(zB, 64, e3W, e3b, e3Ws, e3bs, ybuf, zA);

  // grid-sample + concat -> f16 feature matrix
  gather_kernel<<<BN / 256, 256, 0, stream>>>(f0, f1, f2, f3, WH, zA, feats);

  // dec1 weights -> f16 [128,1024] transposed (W | Ws)
  pack_w_kernel<<<(KDIM * 128) / 256, 256, 0, stream>>>(d1W, d1b, d1Ws, d1bs, wcat_t, bcat);

  // dec1 GEMM (WMMA, TDM double-buffered) + fused norm statistics
  gemm_wmma_kernel<<<BN / 128, 256, 0, stream>>>(feats, wcat_t, bcat, ybuf, stats);

  // dec1 InstanceNorm + ReLU + shortcut
  dec1_norm_kernel<<<(BN * 64) / 256, 256, 0, stream>>>(ybuf, stats, zB);

  // decoder tail: 64 -> 32 -> 16
  small_layer_kernel<64, 32><<<BATCH, 1024, 0, stream>>>(zB, 64, d2W, d2b, d2Ws, d2bs, ybuf, zA);
  small_layer_kernel<32, 16><<<BATCH, 1024, 0, stream>>>(zA, 64, d3W, d3b, d3Ws, d3bs, ybuf, zB);

  // genz + per-batch mean centering + output assembly
  final_kernel<<<BATCH, 1024, 0, stream>>>(zB, WH, gzW, gzb, out, ybuf);
}
